// TransformerBlock_3702261809285
// MI455X (gfx1250) — matrix operand order repaired and checked by reading, not compiled, not run on hardware
//
#include <hip/hip_runtime.h>

// ---------------------------------------------------------------------------
// Transformer block for MI455X (gfx1250, wave32, WMMA bf16 -> f32)
// ---------------------------------------------------------------------------
typedef __bf16 bf16;
typedef __attribute__((ext_vector_type(16))) __bf16 v16bf;
typedef __attribute__((ext_vector_type(8)))  __bf16 v8bf;
typedef __attribute__((ext_vector_type(8)))  float  v8f;
typedef __attribute__((ext_vector_type(4)))  int    v4i;

// builtin prototype (from hipcc diagnostic): (v4i AS1* src, v4i AS3* dst, Ii, Ii)
typedef __attribute__((address_space(1))) v4i* gv4i_p;
typedef __attribute__((address_space(3))) v4i* lv4i_p;

constexpr int SEQ  = 2048;
constexpr int DIM  = 1024;
constexpr int NH   = 16;
constexpr int DH   = 64;
constexpr int DFF  = 4096;
constexpr int NCB  = 4;
constexpr int CDIM = 64;

// ---- gfx1250 async global->LDS copy (16B per lane), ASYNCcnt tracked ------
__device__ inline void async_copy16(const bf16* g, bf16* l) {
#if __has_builtin(__builtin_amdgcn_global_load_async_to_lds_b128)
  __builtin_amdgcn_global_load_async_to_lds_b128(
      (gv4i_p)(unsigned long long)g,
      (lv4i_p)(unsigned int)(unsigned long long)l, 0, 0);
#else
  *(v8bf*)l = *(const v8bf*)g;   // synchronous fallback
#endif
}
template <int N> __device__ inline void wait_async() {
#if __has_builtin(__builtin_amdgcn_global_load_async_to_lds_b128)
#if __has_builtin(__builtin_amdgcn_s_wait_asynccnt)
  __builtin_amdgcn_s_wait_asynccnt(N);
#else
  if (N == 0) asm volatile("s_wait_asynccnt 0x0" ::: "memory");
  else        asm volatile("s_wait_asynccnt 0x1" ::: "memory");
#endif
#endif
}

// ---- WMMA operand loaders (contiguous k = 16*hi + s, same in A and B, 16x16x32 bf16) -------------
// A (16xK=32 tile): lane l holds row m=l&15; lanes 0-15: K 0..15,
// lanes 16-31: K 16..31.
__device__ inline v16bf load_a_tile(const bf16* __restrict__ A, int lda,
                                    int m0, int kk, int lane) {
  int row = lane & 15, hi = lane >> 4;
  const bf16* p = A + (size_t)(m0 + row) * lda + kk + hi * 16;
  union { v16bf v; v8bf h[2]; } u;
  u.h[0] = *(const v8bf*)(p);
  u.h[1] = *(const v8bf*)(p + 8);
  return u.v;
}
// B (K=32 x 16 tile), NT gemm: B[k,n] = W[n0+n][kk+k]; lane l holds column
// n=l&15; lanes 0-15: K 0..15, lanes 16-31: K 16..31 (contiguous 32B load).
__device__ inline v16bf load_b_tile(const bf16* __restrict__ B, int ldb,
                                    int n0, int kk, int lane) {
  int col = lane & 15, hi = lane >> 4;
  return *(const v16bf*)(B + (size_t)(n0 + col) * ldb + kk + hi * 16);
}

#define WMMA_BF16(A_, B_, C_) \
  __builtin_amdgcn_wmma_f32_16x16x32_bf16(false, (A_), false, (B_), (short)0, (C_), false, false)

// ---------------------------------------------------------------------------
// NT GEMM with block-level LDS staging of the B panel via async-to-LDS.
// Block = 256 threads = 8 waves -> 128x64 macro tile (wave w: rows m0+16w).
// B panel (64 cols x 32 k, bf16 = 4 KB) double-buffered in LDS; all 8 waves
// consume it with ds_load_b128 while the next panel streams in (ASYNCcnt).
// ---------------------------------------------------------------------------
__global__ __launch_bounds__(256) void gemm_nt_kernel(
    const bf16* __restrict__ A, int lda,
    const bf16* __restrict__ B, int ldb,
    float* __restrict__ C, int ldc,
    const float* __restrict__ resid,
    int M, int N, int K) {
  __shared__ __align__(32) bf16 Bs[2][64 * 32];
  const int tid = threadIdx.x;
  const int wv = tid >> 5, lane = tid & 31;
  const int mb = blockIdx.x % (M >> 7);
  const int nb = blockIdx.x / (M >> 7);
  const int m0 = (mb << 7) + (wv << 4);
  const int n0 = nb << 6;
  const int srow = tid >> 2, sseg = tid & 3;   // staging: 16B per thread

  // stage first panel
  async_copy16(B + (size_t)(n0 + srow) * ldb + 0 + sseg * 8,
               &Bs[0][srow * 32 + sseg * 8]);

  v8f acc[4] = {};
  int buf = 0;
  for (int kk = 0; kk < K; kk += 32) {
    const bool has_next = (kk + 32 < K);
    if (has_next)
      async_copy16(B + (size_t)(n0 + srow) * ldb + (kk + 32) + sseg * 8,
                   &Bs[buf ^ 1][srow * 32 + sseg * 8]);
    if (has_next) wait_async<1>(); else wait_async<0>();
    __syncthreads();                       // current panel valid block-wide

    v16bf a = load_a_tile(A, lda, m0, kk, lane);
    if (has_next)
      __builtin_prefetch(A + (size_t)(m0 + (lane & 15)) * lda + kk + 32, 0, 3);
    const int col = lane & 15, hi = lane >> 4;
#pragma unroll
    for (int nn = 0; nn < 4; ++nn) {
      v16bf b = *(const v16bf*)(&Bs[buf][(nn * 16 + col) * 32 + hi * 16]);
      acc[nn] = WMMA_BF16(a, b, acc[nn]);
    }
    __syncthreads();                       // all waves done reading this panel
    buf ^= 1;
  }

  const int col = lane & 15, hi = lane >> 4;
#pragma unroll
  for (int nn = 0; nn < 4; ++nn)
#pragma unroll
    for (int r = 0; r < 8; ++r) {
      int m = m0 + r + 8 * hi, n = n0 + nn * 16 + col;
      float v = acc[nn][r];
      if (resid) v += resid[(size_t)m * ldc + n];
      C[(size_t)m * ldc + n] = v;
    }
}

// ---------------------------------------------------------------------------
// cache_linear "up" with in-register product over ncb code-books.
// Adown: bf16 [SEQ][NCB*CDIM]; Wup: bf16 [NCB][N][CDIM]; P: f32 [SEQ][N]
// ---------------------------------------------------------------------------
__global__ __launch_bounds__(256) void cache_up_prod_kernel(
    const bf16* __restrict__ Adown, const bf16* __restrict__ Wup,
    float* __restrict__ P, int N) {
  int wave = (blockIdx.x * 256 + threadIdx.x) >> 5;
  int lane = threadIdx.x & 31;
  int mt = SEQ >> 4, nt = N >> 6;
  if (wave >= mt * nt) return;
  int m0 = (wave % mt) << 4;
  int n0 = (wave / mt) << 6;
  v8f prod[4];
#pragma unroll
  for (int nn = 0; nn < 4; ++nn)
#pragma unroll
    for (int r = 0; r < 8; ++r) prod[nn][r] = 1.0f;
  for (int h = 0; h < NCB; ++h) {
    v8f acc[4] = {};
#pragma unroll
    for (int kk = 0; kk < CDIM; kk += 32) {
      v16bf a = load_a_tile(Adown, NCB * CDIM, m0, h * CDIM + kk, lane);
#pragma unroll
      for (int nn = 0; nn < 4; ++nn) {
        v16bf b = load_b_tile(Wup + (size_t)h * N * CDIM, CDIM, n0 + nn * 16, kk, lane);
        acc[nn] = WMMA_BF16(a, b, acc[nn]);
      }
    }
#pragma unroll
    for (int nn = 0; nn < 4; ++nn)
#pragma unroll
      for (int r = 0; r < 8; ++r) prod[nn][r] *= acc[nn][r];
  }
  int col = lane & 15, hi = lane >> 4;
#pragma unroll
  for (int nn = 0; nn < 4; ++nn)
#pragma unroll
    for (int r = 0; r < 8; ++r)
      P[(size_t)(m0 + r + 8 * hi) * N + n0 + nn * 16 + col] = prod[nn][r];
}

// ---------------------------------------------------------------------------
// Flash-style causal attention. One wave per (head, 16-query-row tile).
// Q,K: bf16 [NH][SEQ][DH]; Vt: bf16 [NH][DH][SEQ]; O: bf16 [SEQ][DIM]
// ---------------------------------------------------------------------------
__global__ __launch_bounds__(128) void attention_kernel(
    const bf16* __restrict__ Q, const bf16* __restrict__ Kc,
    const bf16* __restrict__ Vt, bf16* __restrict__ O) {
  __shared__ float pbuf[4][16 * 32];
  const int wv = threadIdx.x >> 5;
  const int lane = threadIdx.x & 31;
  const int wave = blockIdx.x * 4 + wv;
  const int head = wave / (SEQ / 16);
  const int t0 = (wave % (SEQ / 16)) * 16;
  const int col = lane & 15, hi = lane >> 4;

  const bf16* Qh = Q + (size_t)head * SEQ * DH;
  const bf16* Kh = Kc + (size_t)head * SEQ * DH;
  const bf16* Vh = Vt + (size_t)head * DH * SEQ;

  v16bf qa[2];
#pragma unroll
  for (int c = 0; c < 2; ++c) qa[c] = load_a_tile(Qh, DH, t0, c * 32, lane);

  v8f oacc[4] = {};
  float Mrow[8], Lrow[8];
#pragma unroll
  for (int r = 0; r < 8; ++r) { Mrow[r] = -__builtin_inff(); Lrow[r] = 0.f; }

  for (int kk = 0; kk <= t0 + 15; kk += 32) {
    v8f s0 = {}, s1 = {};
#pragma unroll
    for (int c = 0; c < 2; ++c) {   // dh split: K = 2 x 32
      v16bf bk0 = load_b_tile(Kh, DH, kk, c * 32, lane);
      v16bf bk1 = load_b_tile(Kh, DH, kk + 16, c * 32, lane);
      s0 = WMMA_BF16(qa[c], bk0, s0);
      s1 = WMMA_BF16(qa[c], bk1, s1);
    }
    float scl[8];
#pragma unroll
    for (int r = 0; r < 8; ++r) {
      int m = t0 + r + 8 * hi;               // global query row of this slot
      float a0 = s0[r] * 0.125f;             // 1/sqrt(64)
      float a1 = s1[r] * 0.125f;
      if (kk + col > m)      a0 = -__builtin_inff();   // causal mask
      if (kk + 16 + col > m) a1 = -__builtin_inff();
      float rm = fmaxf(a0, a1);              // row-max across 16 lanes of half
      rm = fmaxf(rm, __shfl_xor(rm, 1, 32));
      rm = fmaxf(rm, __shfl_xor(rm, 2, 32));
      rm = fmaxf(rm, __shfl_xor(rm, 4, 32));
      rm = fmaxf(rm, __shfl_xor(rm, 8, 32));
      float mnew = fmaxf(Mrow[r], rm);
      scl[r] = __expf(Mrow[r] - mnew);
      float p0 = __expf(a0 - mnew);
      float p1 = __expf(a1 - mnew);
      float rs = p0 + p1;                    // row-sum
      rs += __shfl_xor(rs, 1, 32);
      rs += __shfl_xor(rs, 2, 32);
      rs += __shfl_xor(rs, 4, 32);
      rs += __shfl_xor(rs, 8, 32);
      Lrow[r] = Lrow[r] * scl[r] + rs;
      Mrow[r] = mnew;
      // dump P (D-layout) to LDS row-major [row][key]
      pbuf[wv][(r + 8 * hi) * 32 + col] = p0;
      pbuf[wv][(r + 8 * hi) * 32 + 16 + col] = p1;
    }
#pragma unroll
    for (int nn = 0; nn < 4; ++nn)
#pragma unroll
      for (int r = 0; r < 8; ++r) oacc[nn][r] *= scl[r];
    asm volatile("s_wait_dscnt 0" ::: "memory");
    // reload P in WMMA A-layout (K=32) and convert to bf16
    const float* pr = &pbuf[wv][col * 32];
    v16bf pa;
#pragma unroll
    for (int j = 0; j < 8; ++j) {
      pa[j]     = (bf16)pr[hi * 16 + j];
      pa[j + 8] = (bf16)pr[hi * 16 + 8 + j];
    }
#pragma unroll
    for (int nn = 0; nn < 4; ++nn) {       // O += P * V  (dh split into 4 N-tiles)
      v16bf bv = load_b_tile(Vh, SEQ, nn * 16, kk, lane);
      oacc[nn] = WMMA_BF16(pa, bv, oacc[nn]);
    }
  }
#pragma unroll
  for (int nn = 0; nn < 4; ++nn)
#pragma unroll
    for (int r = 0; r < 8; ++r) {
      float v = oacc[nn][r] / Lrow[r];
      O[(size_t)(t0 + r + 8 * hi) * DIM + head * DH + nn * 16 + col] = (bf16)v;
    }
}

// ---------------------------------------------------------------------------
// RoPE + pack: qf/kf f32 [SEQ][DIM] -> bf16 [NH][SEQ][DH]; v -> bf16 [NH][DH][SEQ]
// ---------------------------------------------------------------------------
__global__ void rope_pack_kernel(const float* __restrict__ qf,
                                 const float* __restrict__ kf,
                                 const float* __restrict__ vf,
                                 bf16* __restrict__ qb, bf16* __restrict__ kb,
                                 bf16* __restrict__ vT) {
  int idx = blockIdx.x * blockDim.x + threadIdx.x;
  if (idx >= SEQ * DIM) return;
  int t = idx / DIM, d = idx % DIM;
  int h = d / DH, dd = d % DH;
  vT[((size_t)h * DH + dd) * SEQ + t] = (bf16)vf[idx];
  if ((dd & 1) == 0) {
    int p = dd >> 1;
    // theta^{-2p/dh} = exp(-(2p/dh)*ln(10000))
    float inv = __expf(-((float)(2 * p) / (float)DH) * 9.210340372f);
    float ang = (float)t * inv;
    float c = __cosf(ang), s = __sinf(ang);
    size_t base = ((size_t)h * SEQ + t) * DH + dd;
    float q1 = qf[idx], q2 = qf[idx + 1];
    qb[base]     = (bf16)(q1 * c - q2 * s);
    qb[base + 1] = (bf16)(q1 * s + q2 * c);
    float k1 = kf[idx], k2 = kf[idx + 1];
    kb[base]     = (bf16)(k1 * c - k2 * s);
    kb[base + 1] = (bf16)(k1 * s + k2 * c);
  }
}

// ---------------------------------------------------------------------------
// RMS over rows + post-op. mode 0: out_bf16 = rms(x)*W ; mode 1: out_bf16 =
// silu(rms(x)) ; mode 2: out_f32 = resid + rms(x)
// ---------------------------------------------------------------------------
__global__ __launch_bounds__(256) void rms_post_kernel(
    const float* __restrict__ X, const float* __restrict__ W,
    const float* __restrict__ resid, bf16* __restrict__ outb,
    float* __restrict__ outf, int N, int mode) {
  int t = blockIdx.x;
  const float* xr = X + (size_t)t * N;
  float ss = 0.f;
  for (int i = threadIdx.x; i < N; i += 256) { float v = xr[i]; ss += v * v; }
#pragma unroll
  for (int m = 16; m >= 1; m >>= 1) ss += __shfl_xor(ss, m, 32);
  __shared__ float red[8];
  if ((threadIdx.x & 31) == 0) red[threadIdx.x >> 5] = ss;
  __syncthreads();
  float tot = 0.f;
#pragma unroll
  for (int w = 0; w < 8; ++w) tot += red[w];
  float scale = rsqrtf(tot / (float)N + 1e-6f);
  for (int i = threadIdx.x; i < N; i += 256) {
    float v = xr[i] * scale;
    if (mode == 0) {
      outb[(size_t)t * N + i] = (bf16)(v * (W ? W[i] : 1.f));
    } else if (mode == 1) {
      float s = v / (1.f + __expf(-v));
      outb[(size_t)t * N + i] = (bf16)s;
    } else {
      outf[(size_t)t * N + i] = resid[(size_t)t * N + i] + v;
    }
  }
}

// ---------------------------------------------------------------------------
// Weight conversion helpers
// ---------------------------------------------------------------------------
__global__ void f32_to_bf16_kernel(const float* __restrict__ in,
                                   bf16* __restrict__ out, int n) {
  int i = blockIdx.x * blockDim.x + threadIdx.x;
  if (i < n) out[i] = (bf16)in[i];
}
// in: f32 [NCB][K][CDIM] -> out: bf16 [NCB*CDIM][K] (NT-GEMM B layout)
__global__ void transpose_down_kernel(const float* __restrict__ in,
                                      bf16* __restrict__ out, int K) {
  int i = blockIdx.x * blockDim.x + threadIdx.x;
  if (i >= NCB * K * CDIM) return;
  int c = i % CDIM;
  int k = (i / CDIM) % K;
  int h = i / (CDIM * K);
  out[((size_t)h * CDIM + c) * K + k] = (bf16)in[i];
}

// ---------------------------------------------------------------------------
static inline int cdiv(int a, int b) { return (a + b - 1) / b; }

extern "C" void kernel_launch(void* const* d_in, const int* in_sizes, int n_in,
                              void* d_out, int out_size, void* d_ws, size_t ws_size,
                              hipStream_t stream) {
  (void)in_sizes; (void)n_in; (void)out_size; (void)ws_size;
  const float* x   = (const float*)d_in[0];
  const float* ln1 = (const float*)d_in[1];
  const float* ln2 = (const float*)d_in[2];
  const float* wq  = (const float*)d_in[3];
  const float* wk  = (const float*)d_in[4];
  const float* wv  = (const float*)d_in[5];
  const float* wo  = (const float*)d_in[6];
  const float* w1d = (const float*)d_in[7];
  const float* w1u = (const float*)d_in[8];
  const float* w2d = (const float*)d_in[9];
  const float* w2u = (const float*)d_in[10];
  float* out = (float*)d_out;
  char* ws = (char*)d_ws;
  const size_t MB = 1024 * 1024;

  // ---- workspace layout (high-water ~92 MB, with region reuse) ----
  bf16* wq_b  = (bf16*)(ws + 0 * MB);
  bf16* wk_b  = (bf16*)(ws + 2 * MB);
  bf16* wv_b  = (bf16*)(ws + 4 * MB);
  bf16* wo_b  = (bf16*)(ws + 6 * MB);
  bf16* w1u_b = (bf16*)(ws + 8 * MB);    // 2 MB
  bf16* w2u_b = (bf16*)(ws + 10 * MB);   // 0.5 MB
  bf16* w1dt  = (bf16*)(ws + 11 * MB);   // 0.5 MB
  bf16* w2dt  = (bf16*)(ws + 12 * MB);   // 2 MB
  bf16*  h_b  = (bf16*)(ws + 16 * MB);   // 4 MB
  float* qf   = (float*)(ws + 20 * MB);  // 8 MB
  float* kf   = (float*)(ws + 28 * MB);  // 8 MB
  float* vf   = (float*)(ws + 36 * MB);  // 8 MB
  bf16*  qb   = (bf16*)(ws + 44 * MB);   // 4 MB
  bf16*  kb   = (bf16*)(ws + 48 * MB);   // 4 MB
  bf16*  vtb  = (bf16*)(ws + 52 * MB);   // 4 MB
  bf16*  ob   = (bf16*)(ws + 56 * MB);   // 4 MB
  float* x2   = (float*)(ws + 60 * MB);  // 8 MB (live to end)
  bf16*  h2b  = (bf16*)(ws + 68 * MB);   // 4 MB
  float* ad   = (float*)(ws + 72 * MB);  // 2 MB
  bf16*  adb  = (bf16*)(ws + 74 * MB);   // 1 MB
  float* prod = (float*)(ws + 16 * MB);  // 32 MB, reuses h_b/qf/kf (dead)
  bf16*  silu = (bf16*)(ws + 76 * MB);   // 16 MB -> ends 92 MB
  float* bd   = (float*)(ws + 44 * MB);  // 2 MB, reuses qb (dead)
  bf16*  bdb  = (bf16*)(ws + 46 * MB);   // 1 MB
  float* prod2= (float*)(ws + 48 * MB);  // 8 MB, reuses kb/vtb (dead)

  // ---- 1. weight conversion -------------------------------------------------
  f32_to_bf16_kernel<<<cdiv(DIM * DIM, 256), 256, 0, stream>>>(wq, wq_b, DIM * DIM);
  f32_to_bf16_kernel<<<cdiv(DIM * DIM, 256), 256, 0, stream>>>(wk, wk_b, DIM * DIM);
  f32_to_bf16_kernel<<<cdiv(DIM * DIM, 256), 256, 0, stream>>>(wv, wv_b, DIM * DIM);
  f32_to_bf16_kernel<<<cdiv(DIM * DIM, 256), 256, 0, stream>>>(wo, wo_b, DIM * DIM);
  f32_to_bf16_kernel<<<cdiv(NCB * DFF * CDIM, 256), 256, 0, stream>>>(w1u, w1u_b, NCB * DFF * CDIM);
  f32_to_bf16_kernel<<<cdiv(NCB * DIM * CDIM, 256), 256, 0, stream>>>(w2u, w2u_b, NCB * DIM * CDIM);
  transpose_down_kernel<<<cdiv(NCB * DIM * CDIM, 256), 256, 0, stream>>>(w1d, w1dt, DIM);
  transpose_down_kernel<<<cdiv(NCB * DFF * CDIM, 256), 256, 0, stream>>>(w2d, w2dt, DFF);

  // ---- 2. h = rms(x)*ln1 (bf16) --------------------------------------------
  rms_post_kernel<<<SEQ, 256, 0, stream>>>(x, ln1, nullptr, h_b, nullptr, DIM, 0);

  // ---- 3. QKV projections (LDS-staged NT WMMA GEMMs) -----------------------
  int gqkv = (SEQ / 128) * (DIM / 64);
  gemm_nt_kernel<<<gqkv, 256, 0, stream>>>(h_b, DIM, wq_b, DIM, qf, DIM, nullptr, SEQ, DIM, DIM);
  gemm_nt_kernel<<<gqkv, 256, 0, stream>>>(h_b, DIM, wk_b, DIM, kf, DIM, nullptr, SEQ, DIM, DIM);
  gemm_nt_kernel<<<gqkv, 256, 0, stream>>>(h_b, DIM, wv_b, DIM, vf, DIM, nullptr, SEQ, DIM, DIM);

  // ---- 4. RoPE + head-major bf16 pack (V transposed) -----------------------
  rope_pack_kernel<<<cdiv(SEQ * DIM, 256), 256, 0, stream>>>(qf, kf, vf, qb, kb, vtb);

  // ---- 5. causal flash attention -------------------------------------------
  attention_kernel<<<(NH * (SEQ / 16)) / 4, 128, 0, stream>>>(qb, kb, vtb, ob);

  // ---- 6. out projection + residual: x2 = x + o @ wo^T ---------------------
  gemm_nt_kernel<<<gqkv, 256, 0, stream>>>(ob, DIM, wo_b, DIM, x2, DIM, x, SEQ, DIM, DIM);

  // ---- 7. h2 = rms(x2)*ln2 --------------------------------------------------
  rms_post_kernel<<<SEQ, 256, 0, stream>>>(x2, ln2, nullptr, h2b, nullptr, DIM, 0);

  // ---- 8. cache_linear #1: down, up+prod, rms+silu -------------------------
  gemm_nt_kernel<<<(SEQ / 128) * ((NCB * CDIM) / 64), 256, 0, stream>>>(
      h2b, DIM, w1dt, DIM, ad, NCB * CDIM, nullptr, SEQ, NCB * CDIM, DIM);
  f32_to_bf16_kernel<<<cdiv(SEQ * NCB * CDIM, 256), 256, 0, stream>>>(ad, adb, SEQ * NCB * CDIM);
  cache_up_prod_kernel<<<cdiv((SEQ / 16) * (DFF / 64), 8), 256, 0, stream>>>(adb, w1u_b, prod, DFF);
  rms_post_kernel<<<SEQ, 256, 0, stream>>>(prod, nullptr, nullptr, silu, nullptr, DFF, 1);

  // ---- 9. cache_linear #2: down, up+prod, rms + final residual -------------
  gemm_nt_kernel<<<(SEQ / 128) * ((NCB * CDIM) / 64), 256, 0, stream>>>(
      silu, DFF, w2dt, DFF, bd, NCB * CDIM, nullptr, SEQ, NCB * CDIM, DFF);
  f32_to_bf16_kernel<<<cdiv(SEQ * NCB * CDIM, 256), 256, 0, stream>>>(bd, bdb, SEQ * NCB * CDIM);
  cache_up_prod_kernel<<<cdiv((SEQ / 16) * (DIM / 64), 8), 256, 0, stream>>>(bdb, w2u_b, prod2, DIM);
  rms_post_kernel<<<SEQ, 256, 0, stream>>>(prod2, nullptr, x2, nullptr, out, DIM, 2);
}
